// KNNBCPolicy_34170759807372
// MI455X (gfx1250) — matrix operand order repaired and checked by reading, not compiled, not run on hardware
//
#include <hip/hip_runtime.h>

// ---------------------------------------------------------------------------
// KNN behavior-cloning policy for MI455X (gfx1250, wave32, WMMA).
//
// score[b,n] = 0.5*||bank_n||^2 - <obs_b, bank_n>   (same ranking as cdist^2;
// the per-row ||obs_b||^2 term is constant under per-row top-k and dropped).
//
// GEMM runs in bf16 with f32 accumulation via v_wmma_f32_16x16x32_bf16.
// bf16 bank copy (102 MB) is L2-resident (192 MB L2), so the 16 M-block
// passes stream from L2; HBM sees the f32 bank only once (prep kernel).
// ---------------------------------------------------------------------------

typedef __attribute__((ext_vector_type(16))) __bf16 v16bf;
typedef __attribute__((ext_vector_type(8)))  float  v8f;

#define N_BANK   100000
#define DIM      512
#define BQ       1024
#define ADIM     32
#define KNN      16
#define NCHUNKS  25
#define CHUNK    4032            // 25*4032 = 100800 >= 100000, multiple of 64
#define ITERS    63              // CHUNK / 64
#define MGRID    16              // 1024 / 64 queries per workgroup
#define NCAND    (NCHUNKS * KNN) // 400 candidates per query at finalize

#define INF __builtin_inff()

// workspace layout (bytes); total required ~107.2 MB
#define OFF_BANKBF 0ull          // 100000*512*2 = 102,400,000
#define OFF_OBSBF  102400000ull  // 1024*512*2   =   1,048,576
#define OFF_B2     103448576ull  // 100000*4     =     400,000 (+pad)
#define OFF_PS     103848704ull  // 1024*25*16*4 =   1,638,400
#define OFF_PI     105487104ull  // 1024*25*16*4 =   1,638,400

static __device__ __forceinline__ unsigned short f2bf(float f) {
  unsigned u = __float_as_uint(f);
  u += 0x7FFFu + ((u >> 16) & 1u);          // round to nearest even
  return (unsigned short)(u >> 16);
}

// ---------------------------------------------------------------------------
// Kernel 1: f32 rows -> bf16 rows (+ optional 0.5*||row||^2).
// One wave32 per 512-dim row; 8 rows per 256-thread block.
// ---------------------------------------------------------------------------
__global__ void prep_rows(const float* __restrict__ src,
                          unsigned short* __restrict__ dst,
                          float* __restrict__ b2h, int nrows) {
  const int lane = threadIdx.x & 31;
  const int wv   = threadIdx.x >> 5;
  const int row  = blockIdx.x * 8 + wv;
  if (row >= nrows) return;

  const float4* s4 = reinterpret_cast<const float4*>(src + (size_t)row * DIM + lane * 16);
  float4 v0 = s4[0], v1 = s4[1], v2 = s4[2], v3 = s4[3];
  float f[16] = {v0.x, v0.y, v0.z, v0.w, v1.x, v1.y, v1.z, v1.w,
                 v2.x, v2.y, v2.z, v2.w, v3.x, v3.y, v3.z, v3.w};

  float sum = 0.f;
  unsigned o[8];
#pragma unroll
  for (int i = 0; i < 8; ++i) {
    sum += f[2*i] * f[2*i] + f[2*i+1] * f[2*i+1];
    o[i] = (unsigned)f2bf(f[2*i]) | ((unsigned)f2bf(f[2*i+1]) << 16);
  }

  uint4* d4 = reinterpret_cast<uint4*>(dst + (size_t)row * DIM + lane * 16);
  d4[0] = make_uint4(o[0], o[1], o[2], o[3]);
  d4[1] = make_uint4(o[4], o[5], o[6], o[7]);

  if (b2h) {
#pragma unroll
    for (int m = 16; m >= 1; m >>= 1) sum += __shfl_xor(sum, m, 32);
    if (lane == 0) b2h[row] = 0.5f * sum;
  }
}

// ---------------------------------------------------------------------------
// Kernel 2: fused bf16-WMMA score GEMM + per-chunk top-16.
// Block = 128 threads = 4 waves. Each wave owns 16 queries (A panel held in
// 128 VGPRs across 16 kb-fragments). Per iteration: 4 n-tiles x 16 WMMAs,
// scores dumped to LDS, 2 threads/query maintain sorted top-16 in registers.
// ---------------------------------------------------------------------------
__global__ void __launch_bounds__(128)
knn_gemm_topk(const unsigned short* __restrict__ obsBf,
              const unsigned short* __restrict__ bankBf,
              const float* __restrict__ b2h,
              float* __restrict__ pS, int* __restrict__ pI) {
  __shared__ float s_scores[64 * 65];   // 64 queries x 64 cols, stride 65 (bank-conflict pad)
  __shared__ float s_candS[64 * 32];
  __shared__ int   s_candI[64 * 32];

  const int tid  = threadIdx.x;
  const int lane = tid & 31;
  const int wv   = tid >> 5;       // 0..3
  const int hi   = lane >> 4;      // 0/1 half-wave
  const int l15  = lane & 15;
  const int mbase  = blockIdx.x * 64;
  const int nstart = blockIdx.y * CHUNK;

  // ---- preload A fragments: 16 queries of this wave, all of D=512 ----
  // ISA 16-bit A 16x32 layout: lane m=l15; elems 0..7 -> dims base+8*hi,
  // elems 8..15 -> dims base+16+8*hi.
  v16bf afrag[16];
  {
    const unsigned short* arow = obsBf + (size_t)(mbase + wv * 16 + l15) * DIM;
#pragma unroll
    for (int kb = 0; kb < 16; ++kb) {
      union { uint4 u[2]; v16bf v; } t;
      t.u[0] = *reinterpret_cast<const uint4*>(arow + kb * 32 + 8 * hi);
      t.u[1] = *reinterpret_cast<const uint4*>(arow + kb * 32 + 16 + 8 * hi);
      afrag[kb] = t.v;
    }
  }

  // per-thread sorted top-16 (ascending; [15] = worst). All static indices.
  float sc[16]; int ix[16];
#pragma unroll
  for (int i = 0; i < 16; ++i) { sc[i] = INF; ix[i] = 0; }
  const int q = tid >> 1, sub = tid & 1;

  for (int it = 0; it < ITERS; ++it) {
    const int nbase = nstart + it * 64;
#pragma unroll
    for (int nt = 0; nt < 4; ++nt) {
      const int nrow = nbase + nt * 16 + l15;
      const int ncl  = nrow < N_BANK ? nrow : N_BANK - 1;
      const unsigned short* brow = bankBf + (size_t)ncl * DIM;
      // pull next 64-column step toward the caches
      const int npf = (ncl + 64 < N_BANK) ? ncl + 64 : ncl;
      __builtin_prefetch(bankBf + (size_t)npf * DIM + 16 * hi, 0, 1);

      // ISA 16-bit B 32x16 layout: lane col n=l15; elems 0..7 -> dims
      // kb*32+8*hi, elems 8..15 -> dims kb*32+16+8*hi (matches A).
      v8f c = {};
#pragma unroll
      for (int kb = 0; kb < 16; ++kb) {
        union { uint4 u[2]; v16bf v; } t;
        const uint4* p = reinterpret_cast<const uint4*>(brow + kb * 32 + 8 * hi);
        t.u[0] = p[0];
        t.u[1] = p[2];
        c = __builtin_amdgcn_wmma_f32_16x16x32_bf16(
                false, afrag[kb], false, t.v, (short)0, c, false, false);
      }
      const float bb    = b2h[ncl];
      const bool  valid = nrow < N_BANK;
      // C layout: reg r holds (m = r + 8*hi, n = l15)
#pragma unroll
      for (int r = 0; r < 8; ++r) {
        const float s = valid ? (bb - c[r]) : INF;
        s_scores[(wv * 16 + r + 8 * hi) * 65 + nt * 16 + l15] = s;
      }
    }
    __syncthreads();
    // 2 threads per query scan 32 of the 64 columns each
#pragma unroll
    for (int j = 0; j < 32; ++j) {
      const int   col = sub + 2 * j;
      const float s   = s_scores[q * 65 + col];
      if (s < sc[15]) {
        sc[15] = s; ix[15] = nbase + col;
#pragma unroll
        for (int i = 15; i > 0; --i) {
          if (sc[i] < sc[i - 1]) {
            float ts = sc[i]; sc[i] = sc[i - 1]; sc[i - 1] = ts;
            int   ti = ix[i]; ix[i] = ix[i - 1]; ix[i - 1] = ti;
          }
        }
      }
    }
    __syncthreads();
  }

  // ---- chunk-end: merge the two sorted lists per query, spill top-16 ----
#pragma unroll
  for (int j = 0; j < 16; ++j) {
    s_candS[(q * 2 + sub) * 16 + j] = sc[j];
    s_candI[(q * 2 + sub) * 16 + j] = ix[j];
  }
  __syncthreads();
  if (sub == 0) {
    const int b0 = q * 32, b1 = q * 32 + 16;
    int p0 = 0, p1 = 0;
    const size_t obase = ((size_t)(mbase + q) * NCHUNKS + blockIdx.y) * KNN;
    for (int t = 0; t < 16; ++t) {      // p0+p1 = t <= 15, both always < 16
      const float v0 = s_candS[b0 + p0];
      const float v1 = s_candS[b1 + p1];
      if (v0 <= v1) { pS[obase + t] = v0; pI[obase + t] = s_candI[b0 + p0]; ++p0; }
      else          { pS[obase + t] = v1; pI[obase + t] = s_candI[b1 + p1]; ++p1; }
    }
  }
}

// ---------------------------------------------------------------------------
// Kernel 3: per query, reduce 400 chunk candidates -> global top-16, gather
// action rows and mean. One thread per query (trivial cost).
// ---------------------------------------------------------------------------
__global__ void knn_finalize(const float* __restrict__ pS, const int* __restrict__ pI,
                             const float* __restrict__ act, float* __restrict__ out) {
  const int qg = blockIdx.x * blockDim.x + threadIdx.x;   // 0..1023
  float sc[16]; int ix[16];
#pragma unroll
  for (int i = 0; i < 16; ++i) { sc[i] = INF; ix[i] = 0; }
  const size_t base = (size_t)qg * NCAND;
  for (int c = 0; c < NCAND; ++c) {
    const float s = pS[base + c];
    if (s < sc[15]) {
      const int n = pI[base + c];
      sc[15] = s; ix[15] = n;
#pragma unroll
      for (int i = 15; i > 0; --i) {
        if (sc[i] < sc[i - 1]) {
          float ts = sc[i]; sc[i] = sc[i - 1]; sc[i - 1] = ts;
          int   ti = ix[i]; ix[i] = ix[i - 1]; ix[i - 1] = ti;
        }
      }
    }
  }
  float acc[ADIM];
#pragma unroll
  for (int d = 0; d < ADIM; ++d) acc[d] = 0.f;
#pragma unroll
  for (int j = 0; j < 16; ++j) {
    const float* ar = act + (size_t)ix[j] * ADIM;
#pragma unroll
    for (int d = 0; d < ADIM; ++d) acc[d] += ar[d];
  }
#pragma unroll
  for (int d = 0; d < ADIM; ++d) out[(size_t)qg * ADIM + d] = acc[d] * (1.0f / 16.0f);
}

// ---------------------------------------------------------------------------
extern "C" void kernel_launch(void* const* d_in, const int* in_sizes, int n_in,
                              void* d_out, int out_size, void* d_ws, size_t ws_size,
                              hipStream_t stream) {
  const float* obs  = (const float*)d_in[0];   // [1024, 512]
  const float* bank = (const float*)d_in[1];   // [100000, 512]
  const float* act  = (const float*)d_in[2];   // [100000, 32]
  float* out = (float*)d_out;                  // [1024, 32]

  char* ws = (char*)d_ws;                      // requires ~107.2 MB scratch
  unsigned short* bankBf = (unsigned short*)(ws + OFF_BANKBF);
  unsigned short* obsBf  = (unsigned short*)(ws + OFF_OBSBF);
  float* b2h = (float*)(ws + OFF_B2);
  float* pS  = (float*)(ws + OFF_PS);
  int*   pI  = (int*)(ws + OFF_PI);

  prep_rows<<<N_BANK / 8, 256, 0, stream>>>(bank, bankBf, b2h, N_BANK);
  prep_rows<<<BQ / 8,     256, 0, stream>>>(obs,  obsBf,  nullptr, BQ);
  knn_gemm_topk<<<dim3(MGRID, NCHUNKS), 128, 0, stream>>>(obsBf, bankBf, b2h, pS, pI);
  knn_finalize<<<BQ / 256, 256, 0, stream>>>(pS, pI, act, out);
}
